// GRUModel_50861002719326
// MI455X (gfx1250) — matrix operand order repaired and checked by reading, not compiled, not run on hardware
//
#include <hip/hip_runtime.h>
#include <hip/hip_bf16.h>

// ---------------------------------------------------------------------------
// MI455X (gfx1250) persistent-GRU implementation.
//
// Strategy: batch (256) splits into 16 independent slices of 16 rows (one WMMA
// M-tile). Each workgroup (256 threads = 8 wave32) owns one slice and runs the
// entire day(8) x hour(24) recurrence with all state in LDS. GEMMs use
// v_wmma_f32_16x16x32_f16 (f16 operands, f32 accum). Weights are converted
// to f16 once per launch into the workspace.
// ---------------------------------------------------------------------------

typedef _Float16 half_t;
typedef __attribute__((ext_vector_type(16))) _Float16 v16h;
typedef __attribute__((ext_vector_type(8)))  _Float16 v8h;
typedef __attribute__((ext_vector_type(8)))  float    v8f;

#define DEVFN static __device__ __forceinline__

namespace {
constexpr int kB  = 256;   // batch
constexpr int kD  = 8;     // days
constexpr int kNH = 24;    // hours
constexpr int kH  = 128;   // hidden
constexpr int kNG = 384;   // 3*hidden (gates)
constexpr int kMT = 16;    // batch rows per workgroup (one WMMA M tile)

// output section offsets (floats), concat in reference return order
constexpr int OFF_CARP = 0;                          // (256,2)
constexpr int OFF_RTMO = OFF_CARP + kB * 2;          // (256,24,2)
constexpr int OFF_ENBA = OFF_RTMO + kB * kNH * 2;    // (256,24,2)
constexpr int OFF_BICM = OFF_ENBA + kB * kNH * 2;    // (256,24,4)
constexpr int OFF_RTMS = OFF_BICM + kB * kNH * 4;    // (256,24,4)
constexpr int OFF_HID  = OFF_RTMS + kB * kNH * 4;    // 10 x (256,128)
} // namespace

struct KArgs {
  const float* carp_x;
  const float* hx[4];          // rtmo, enba, bicm, rtms inputs (B,D,NH,10)
  const float* h0[5];          // initial hidden states (2,B,128)
  const float* bih[5][2];
  const float* bhh[5][2];
  const float* fcw[5];
  const float* fcb[5];
  const half_t* wihH[5][2];    // f16 weights in workspace (row-major N x Kpad)
  const half_t* whhH[5][2];    // f16 weights (384 x 128)
  float* out;
};

DEVFN float sigmoidf_(float x) { return 1.0f / (1.0f + __expf(-x)); }

// A-operand fragment (16-bit A 16x32 layout, ISA 7.12.2):
// lane<16: elements 0..7 = K k0+0..7, 8..15 = K k0+16..23
// lane>=16: elements 0..7 = K k0+8..15, 8..15 = K k0+24..31
// caller passes p = base + row*ld + k0 + hi*8
DEVFN v16h load_fragA(const half_t* p) {
  v8h lo = *(const v8h*)(p);
  v8h hi = *(const v8h*)(p + 16);
  v16h f;
#pragma unroll
  for (int i = 0; i < 8; ++i) { f[i] = lo[i]; f[i + 8] = hi[i]; }
  return f;
}

// B-operand fragment (K x 16; same K order as A fragment):
// lane<16: elements 0..7 = K k0+0..7, 8..15 = K k0+16..23
// lane>=16: elements 0..7 = K k0+8..15, 8..15 = K k0+24..31
// caller passes p = W_row_ptr + k0 + hi*8  (W row n == B column n)
DEVFN v16h load_fragB(const half_t* p) {
  v8h a = *(const v8h*)(p);
  v8h b = *(const v8h*)(p + 16);
  v16h f;
#pragma unroll
  for (int i = 0; i < 8; ++i) { f[i] = a[i]; f[i + 8] = b[i]; }
  return f;
}

DEVFN v8f wmma_(v16h a, v16h b, v8f c) {
  return __builtin_amdgcn_wmma_f32_16x16x32_f16(false, a, false, b, (short)0, c,
                                                false, false);
}

// one GRU cell: gi = xf16 @ wih^T + bih ; gh = hf16 @ whh^T + bhh ; update hS
// aXp already staged by caller (16 x KX f16, ld=128). Stages hS->aH itself.
template <int KX>
DEVFN void run_cell(int tid, float* hS, const half_t* aXp, half_t* aHp,
                    float* Gi, float* Gh, const half_t* wihH,
                    const half_t* whhH, const float* bih, const float* bhh) {
  for (int i = tid; i < kMT * kH; i += 256) aHp[i] = (half_t)hS[i];
  __syncthreads();

  const int wave = tid >> 5, lane = tid & 31, r = lane & 15, hi = lane >> 4;
  v8f acci[3] = {v8f{}, v8f{}, v8f{}};
  v8f acch[3] = {v8f{}, v8f{}, v8f{}};

#pragma unroll
  for (int kc = 0; kc < 4; ++kc) {
    const int k0 = kc * 32;
    v16h ah = load_fragA(aHp + r * kH + k0 + hi * 8);
#pragma unroll
    for (int t = 0; t < 3; ++t) {
      const int nt = wave * 3 + t;
      v16h bh = load_fragB(whhH + (size_t)(nt * 16 + r) * kH + k0 + hi * 8);
      acch[t] = wmma_(ah, bh, acch[t]);
    }
    if (k0 < KX) {
      v16h ax = load_fragA(aXp + r * kH + k0 + hi * 8);
#pragma unroll
      for (int t = 0; t < 3; ++t) {
        const int nt = wave * 3 + t;
        v16h bi = load_fragB(wihH + (size_t)(nt * 16 + r) * KX + k0 + hi * 8);
        acci[t] = wmma_(ax, bi, acci[t]);
      }
    }
  }

  // epilogue: C element j in lane = (M = j + 8*hi, N = r) of the 16x16 tile
#pragma unroll
  for (int t = 0; t < 3; ++t) {
    const int ncol = (wave * 3 + t) * 16 + r;
    const float bi_ = bih[ncol], bh_ = bhh[ncol];
#pragma unroll
    for (int j = 0; j < 8; ++j) {
      const int m = j + hi * 8;
      Gi[m * kNG + ncol] = acci[t][j] + bi_;
      Gh[m * kNG + ncol] = acch[t][j] + bh_;
    }
  }
  __syncthreads();

  for (int i = tid; i < kMT * kH; i += 256) {
    const int m = i >> 7, c = i & 127;
    const float rg = sigmoidf_(Gi[m * kNG + c] + Gh[m * kNG + c]);
    const float zg = sigmoidf_(Gi[m * kNG + kH + c] + Gh[m * kNG + kH + c]);
    const float ng =
        tanhf(Gi[m * kNG + 2 * kH + c] + rg * Gh[m * kNG + 2 * kH + c]);
    const float hp = hS[i];
    hS[i] = (1.0f - zg) * ng + zg * hp;
  }
  __syncthreads();
}

DEVFN void stage_state(int tid, const float* hS, half_t* aXp) {
  for (int i = tid; i < kMT * kH; i += 256) aXp[i] = (half_t)hS[i];
}

// hourly model input staging with scalar-channel overrides (md: 1=rtmo,
// 2=enba, 3=bicm, 4=rtms). scal: 0=lai 1=fpar 2=rnet 3=lst 4=fqe_u 5=fqe_h
DEVFN void stage_hour_x(int tid, const float* xg, int bs, int day, int hour,
                        int md, const float (*scal)[kMT], half_t* aXp) {
  for (int i = tid; i < kMT * 32; i += 256) {
    const int m = i >> 5, c = i & 31;
    float v = 0.0f;
    if (c < 10)
      v = xg[(((size_t)(bs + m) * kD + day) * kNH + hour) * 10 + c];
    if (c == 0) v = scal[0][m];
    else if (c == 1 && md >= 3) v = scal[1][m];
    else if (c == 3) {
      if (md == 2) v = scal[2][m];
      else if (md == 3) v = scal[3][m];
      else if (md == 4) v = scal[4][m];
    } else if (c == 4 && md == 4) v = scal[5][m];
    aXp[m * kH + c] = (half_t)v;
  }
}

DEVFN void do_fc(int tid, const float* hS, const float* w, const float* b,
                 int outd, float (*fcout)[4]) {
  if (tid < kMT * outd) {
    const int m = tid / outd, o = tid - m * outd;
    float s = b[o];
    const float* wr = w + o * kH;
    const float* hr = hS + m * kH;
#pragma unroll 8
    for (int c = 0; c < kH; ++c) s += hr[c] * wr[c];
    fcout[m][o] = sigmoidf_(s);
  }
}

__global__ __launch_bounds__(256, 1) void gru_forest(KArgs A) {
  __shared__ float  hF[10][kMT][kH];     // 80 KB master f32 hidden states
  __shared__ half_t aX[kMT][kH];         // 4 KB  f16 A operand (input)
  __shared__ half_t aH[kMT][kH];         // 4 KB  f16 A operand (hidden)
  __shared__ float  Gi[kMT][kNG];        // 24 KB gate preacts (input path)
  __shared__ float  Gh[kMT][kNG];        // 24 KB gate preacts (hidden path)
  __shared__ float  scal[6][kMT];
  __shared__ float  fcout[kMT][4];
  __shared__ float  bicmAcc[kMT];
  __shared__ float  bicmDaily[kMT];

  const int tid = threadIdx.x;
  const int bs = blockIdx.x * kMT;

  for (int i = tid; i < 10 * kMT * kH; i += 256) {
    const int st = i >> 11, rem = i & 2047, m = rem >> 7, c = rem & 127;
    hF[st][m][c] = A.h0[st >> 1][(size_t)(st & 1) * kB * kH + (bs + m) * kH + c];
  }
  __syncthreads();

  for (int day = 0; day < kD; ++day) {
    // ---- carp (model 0, states 0/1) ----
    for (int i = tid; i < kMT * 32; i += 256) {
      const int m = i >> 5, c = i & 31;
      float v = 0.0f;
      if (c < 12) v = A.carp_x[((size_t)(bs + m) * kD + day) * 12 + c];
      if (c == 0 && day > 0) v = bicmDaily[m];
      aX[m][c] = (half_t)v;
    }
    __syncthreads();
    run_cell<32>(tid, &hF[0][0][0], &aX[0][0], &aH[0][0], &Gi[0][0], &Gh[0][0],
                 A.wihH[0][0], A.whhH[0][0], A.bih[0][0], A.bhh[0][0]);
    stage_state(tid, &hF[0][0][0], &aX[0][0]);
    __syncthreads();
    run_cell<128>(tid, &hF[1][0][0], &aX[0][0], &aH[0][0], &Gi[0][0], &Gh[0][0],
                  A.wihH[0][1], A.whhH[0][1], A.bih[0][1], A.bhh[0][1]);
    do_fc(tid, &hF[1][0][0], A.fcw[0], A.fcb[0], 2, fcout);
    __syncthreads();
    if (tid < kMT) {
      scal[0][tid] = fcout[tid][0];   // lai
      bicmAcc[tid] = 0.0f;
      if (day == kD - 1) {
        A.out[OFF_CARP + (bs + tid) * 2 + 0] = fcout[tid][0];
        A.out[OFF_CARP + (bs + tid) * 2 + 1] = fcout[tid][1];
      }
    }
    __syncthreads();

    for (int hour = 0; hour < kNH; ++hour) {
      for (int md = 1; md <= 4; ++md) {
        stage_hour_x(tid, A.hx[md - 1], bs, day, hour, md, scal, &aX[0][0]);
        __syncthreads();
        run_cell<32>(tid, &hF[2 * md][0][0], &aX[0][0], &aH[0][0], &Gi[0][0],
                     &Gh[0][0], A.wihH[md][0], A.whhH[md][0], A.bih[md][0],
                     A.bhh[md][0]);
        stage_state(tid, &hF[2 * md][0][0], &aX[0][0]);
        __syncthreads();
        run_cell<128>(tid, &hF[2 * md + 1][0][0], &aX[0][0], &aH[0][0],
                      &Gi[0][0], &Gh[0][0], A.wihH[md][1], A.whhH[md][1],
                      A.bih[md][1], A.bhh[md][1]);
        const int od = (md >= 3) ? 4 : 2;
        const bool needFc = (md != 4) || (day == kD - 1);
        if (needFc)
          do_fc(tid, &hF[2 * md + 1][0][0], A.fcw[md], A.fcb[md], od, fcout);
        __syncthreads();
        if (needFc && tid < kMT) {
          if (md == 1) { scal[1][tid] = fcout[tid][0]; scal[2][tid] = fcout[tid][1]; }
          else if (md == 2) { scal[3][tid] = fcout[tid][0]; }
          else if (md == 3) {
            scal[4][tid] = fcout[tid][2];
            scal[5][tid] = fcout[tid][3];
            bicmAcc[tid] += fcout[tid][0];
          }
          if (day == kD - 1) {
            const int offm = (md == 1) ? OFF_RTMO
                           : (md == 2) ? OFF_ENBA
                           : (md == 3) ? OFF_BICM : OFF_RTMS;
            for (int o = 0; o < od; ++o)
              A.out[offm + ((size_t)(bs + tid) * kNH + hour) * od + o] =
                  fcout[tid][o];
          }
        }
        __syncthreads();
      }
    }
    if (tid < kMT) bicmDaily[tid] = bicmAcc[tid] * (1.0f / 24.0f);
    __syncthreads();
  }

  // final hidden states: ch, roh, eh, bh, rh each (2,B,128)
  for (int i = tid; i < 10 * kMT * kH; i += 256) {
    const int st = i >> 11, rem = i & 2047, m = rem >> 7, c = rem & 127;
    A.out[OFF_HID + (size_t)st * kB * kH + (bs + m) * kH + c] = hF[st][m][c];
  }
}

// ---------------------------------------------------------------------------
// f32 -> f16 weight conversion (layer-0 K zero-padded to 32)
// ---------------------------------------------------------------------------
struct ConvJob { const float* src; half_t* dst; int K; int Kpad; };
struct ConvJobs { ConvJob j[20]; };

__global__ void wconv_kernel(ConvJobs jobs) {
  const ConvJob jb = jobs.j[blockIdx.y];
  const int total = kNG * jb.Kpad;
  for (int i = blockIdx.x * blockDim.x + threadIdx.x; i < total;
       i += gridDim.x * blockDim.x) {
    const int r = i / jb.Kpad, k = i - r * jb.Kpad;
    const float v = (k < jb.K) ? jb.src[r * jb.K + k] : 0.0f;
    jb.dst[i] = (half_t)v;
  }
}

// ---------------------------------------------------------------------------
// d_in layout: [0]=carp_x [1..4]=rtmo/enba/bicm/rtms_x [5..9]=h0 (carp..rtms)
// [10..59]=params leaves in JAX pytree order: top-level keys sorted
// (bicm,carp,enba,rtmo,rtms); per model: cells[0]{bhh,bih,whh,wih},
// cells[1]{...}, fc_b, fc_w.
// ---------------------------------------------------------------------------
extern "C" void kernel_launch(void* const* d_in, const int* in_sizes, int n_in,
                              void* d_out, int out_size, void* d_ws,
                              size_t ws_size, hipStream_t stream) {
  (void)in_sizes; (void)n_in; (void)out_size; (void)ws_size;
  const int l2s[5] = {1, 3, 2, 0, 4};  // logical carp,rtmo,enba,bicm,rtms -> sorted slot
  const int ind[5] = {12, 10, 10, 10, 10};

  KArgs A;
  A.carp_x = (const float*)d_in[0];
  for (int i = 0; i < 4; ++i) A.hx[i] = (const float*)d_in[1 + i];
  for (int i = 0; i < 5; ++i) A.h0[i] = (const float*)d_in[5 + i];

  const float* wih32[5][2];
  const float* whh32[5][2];
  for (int li = 0; li < 5; ++li) {
    const int base = 10 + l2s[li] * 10;
    for (int l = 0; l < 2; ++l) {
      A.bhh[li][l] = (const float*)d_in[base + l * 4 + 0];
      A.bih[li][l] = (const float*)d_in[base + l * 4 + 1];
      whh32[li][l] = (const float*)d_in[base + l * 4 + 2];
      wih32[li][l] = (const float*)d_in[base + l * 4 + 3];
    }
    A.fcb[li] = (const float*)d_in[base + 8];
    A.fcw[li] = (const float*)d_in[base + 9];
  }
  A.out = (float*)d_out;

  half_t* w = (half_t*)d_ws;
  ConvJobs jobs;
  int jn = 0;
  size_t off = 0;
  for (int li = 0; li < 5; ++li) {
    jobs.j[jn++] = {wih32[li][0], w + off, ind[li], 32};
    A.wihH[li][0] = w + off; off += (size_t)kNG * 32;
    jobs.j[jn++] = {whh32[li][0], w + off, kH, kH};
    A.whhH[li][0] = w + off; off += (size_t)kNG * kH;
    jobs.j[jn++] = {wih32[li][1], w + off, kH, kH};
    A.wihH[li][1] = w + off; off += (size_t)kNG * kH;
    jobs.j[jn++] = {whh32[li][1], w + off, kH, kH};
    A.whhH[li][1] = w + off; off += (size_t)kNG * kH;
  }

  wconv_kernel<<<dim3(48, 20), dim3(256), 0, stream>>>(jobs);
  gru_forest<<<dim3(16), dim3(256), 0, stream>>>(A);
}
